// Block_83837761618673
// MI455X (gfx1250) — compile-verified, matrix operand order checked by reading, not run on hardware
//
#include <hip/hip_runtime.h>
#include <hip/hip_bf16.h>

// ---------------------------------------------------------------------------
// Transformer block for MI455X (gfx1250, wave32, WMMA + TDM).
// fp32 in/out; all matmuls run as bf16 WMMA (v_wmma_f32_16x16x32_bf16) with
// fp32 accumulation. Global->LDS tile staging uses the Tensor Data Mover
// (tensor_load_to_lds, TENSORcnt) with double-buffered LDS in the GEMMs.
// ---------------------------------------------------------------------------

typedef __bf16 bf16_t;
typedef __attribute__((ext_vector_type(16))) __bf16 bf16x16;
typedef __attribute__((ext_vector_type(8)))  __bf16 bf16x8;
typedef __attribute__((ext_vector_type(8)))  float  f32x8;
typedef __attribute__((ext_vector_type(4)))  unsigned int u32x4;
typedef __attribute__((ext_vector_type(8)))  int i32x8;
typedef __attribute__((ext_vector_type(4)))  int i32x4;

#define WMMA_BF16(a, b, c) \
  __builtin_amdgcn_wmma_f32_16x16x32_bf16(false, (a), false, (b), (short)0, (c), false, false)

#define BB 4
#define NN 1024
#define DD 1024
#define NHH 16
#define HSS 64
#define HIDD 4096
#define MTOT (BB * NN)   // 4096 rows

// ---------------------------------------------------------------------------
// Tensor Data Mover: 2D tile load Global->LDS with row padding.
// D# packing per CDNA5 ISA (group0 128b, group1 256b).
//   pad_int_code:  pad after 2^(code+1) DWORDs   (tile row)
//   pad_amt_code:  insert (code+1) DWORDs of pad (row stride fixup)
// Issued by one wave; tracked on TENSORcnt. This toolchain's builtin takes
// 6 args (g0, g1, g2, g3, g4, cpol); extra groups zero for 2D tiles.
// ---------------------------------------------------------------------------
__device__ __forceinline__ void tdm_load_2d(const bf16_t* gptr, unsigned lds_addr,
                                            unsigned tensor_d0, unsigned tensor_d1,
                                            unsigned tile_d0, unsigned tile_d1,
                                            unsigned stride0,
                                            unsigned pad_int_code, unsigned pad_amt_code) {
  unsigned long long ga = (unsigned long long)(uintptr_t)gptr;
  u32x4 g0;
  g0[0] = 1u;                                                  // count=1, user D#
  g0[1] = lds_addr;                                            // LDS byte address
  g0[2] = (unsigned)ga;                                        // global_addr[31:0]
  g0[3] = (unsigned)((ga >> 32) & 0x1FFFFFFu) | (2u << 30);    // addr[56:32] | type=2
  i32x8 g1;
  g1[0] = (int)((1u << 16)                      // data_size = 2 bytes (bf16)
              | (1u << 20)                      // pad_enable
              | (pad_int_code << 22)
              | (pad_amt_code << 25));
  g1[1] = (int)((tensor_d0 & 0xFFFFu) << 16);                  // tensor_dim0 lo
  g1[2] = (int)((tensor_d0 >> 16) | ((tensor_d1 & 0xFFFFu) << 16));
  g1[3] = (int)((tensor_d1 >> 16) | (tile_d0 << 16));          // tile_dim0
  g1[4] = (int)tile_d1;                                        // tile_dim1 (tile_dim2=0)
  g1[5] = (int)stride0;                                        // tensor_dim0_stride lo
  g1[6] = 0;
  g1[7] = 0;
  i32x4 g2 = {0, 0, 0, 0};
  i32x4 g3 = {0, 0, 0, 0};
  i32x8 g4 = {0, 0, 0, 0, 0, 0, 0, 0};
  __builtin_amdgcn_tensor_load_to_lds(g0, g1, g2, g3, g4, 0);
}

// Load one 16x32 bf16 operand fragment (A-layout per CDNA5 ISA 7.12.2):
// lane<16 : row=lane,    K = [0..7]  and [16..23]
// lane>=16: row=lane-16, K = [8..15] and [24..31]
// Source is row-major with K contiguous; row stride `ld` elements (16B aligned).
__device__ __forceinline__ bf16x16 load_frag(const bf16_t* base, int ld, int lane) {
  const int row = lane & 15;
  const int k0  = (lane >> 4) << 3;       // 0 or 8
  const bf16_t* p = base + (size_t)row * ld + k0;
  bf16x8 lo = *(const bf16x8*)(p);        // K k0 .. k0+7
  bf16x8 hi = *(const bf16x8*)(p + 16);   // K k0+16 .. k0+23
  bf16x16 r;
#pragma unroll
  for (int i = 0; i < 8; ++i) { r[i] = lo[i]; r[i + 8] = hi[i]; }
  return r;
}

// ---------------------------------------------------------------------------
// fp32 -> bf16 weight conversion
// ---------------------------------------------------------------------------
__global__ __launch_bounds__(256) void cvt_f32_to_bf16(const float* __restrict__ in,
                                                       bf16_t* __restrict__ out, int n) {
  int i = blockIdx.x * 256 + threadIdx.x;
  if (i < n) out[i] = (bf16_t)in[i];
}

// ---------------------------------------------------------------------------
// LayerNorm over D=1024, one workgroup (256 threads) per row, bf16 output.
// ---------------------------------------------------------------------------
__global__ __launch_bounds__(256) void layernorm_to_bf16(const float* __restrict__ x,
                                                         const float* __restrict__ w,
                                                         const float* __restrict__ b,
                                                         bf16_t* __restrict__ out) {
  const int row = blockIdx.x;
  const float* xr = x + (size_t)row * DD;
  float4 v = ((const float4*)xr)[threadIdx.x];          // 256 * 4 = 1024
  float s  = v.x + v.y + v.z + v.w;
  float ss = v.x * v.x + v.y * v.y + v.z * v.z + v.w * v.w;
#pragma unroll
  for (int off = 16; off > 0; off >>= 1) {
    s  += __shfl_xor(s,  off, 32);
    ss += __shfl_xor(ss, off, 32);
  }
  __shared__ float sbuf[8], ssbuf[8];
  if ((threadIdx.x & 31) == 0) { sbuf[threadIdx.x >> 5] = s; ssbuf[threadIdx.x >> 5] = ss; }
  __syncthreads();
  s = 0.f; ss = 0.f;
#pragma unroll
  for (int i = 0; i < 8; ++i) { s += sbuf[i]; ss += ssbuf[i]; }
  const float mu   = s * (1.0f / DD);
  const float var  = ss * (1.0f / DD) - mu * mu;
  const float rstd = rsqrtf(var + 1e-5f);
  const int c = threadIdx.x * 4;
  float4 wv = ((const float4*)w)[threadIdx.x];
  float4 bv = ((const float4*)b)[threadIdx.x];
  bf16_t* o = out + (size_t)row * DD + c;
  o[0] = (bf16_t)((v.x - mu) * rstd * wv.x + bv.x);
  o[1] = (bf16_t)((v.y - mu) * rstd * wv.y + bv.y);
  o[2] = (bf16_t)((v.z - mu) * rstd * wv.z + bv.z);
  o[3] = (bf16_t)((v.w - mu) * rstd * wv.w + bv.w);
}

// ---------------------------------------------------------------------------
// Tiled bf16 WMMA GEMM:  out[M,N] = A[M,K] * W[N,K]^T (+bias)(+GELU)(+res)
// Block 256 threads = 8 waves; block tile 128x128; wave tile 32x64; BK=32.
// TDM double-buffered staging: wave 0 issues tensor_load_to_lds for tile t+1
// while all waves run WMMA on tile t. LDS rows padded to 80B (pad codes 3/3:
// pad 4 DWORDs after every 16 DWORDs) to dodge bank conflicts.
// ---------------------------------------------------------------------------
template <bool HAS_BIAS, bool GELU_ACT, bool RESIDUAL, bool OUT_BF16>
__global__ __launch_bounds__(256) void gemm_bf16_kernel(const bf16_t* __restrict__ A,
                                                        const bf16_t* __restrict__ W,
                                                        const float* __restrict__ bias,
                                                        const float* __restrict__ res,
                                                        void* __restrict__ outp,
                                                        int M, int N, int K) {
  __shared__ __align__(16) bf16_t As[2][128][40];   // 2 x 10KB, 80B row stride
  __shared__ __align__(16) bf16_t Bs[2][128][40];

  const int tid  = threadIdx.x;
  const int lane = tid & 31;
  const int wid  = tid >> 5;
  const int wm   = wid & 3;             // 4 waves along M
  const int wn   = wid >> 2;            // 2 waves along N
  const int m0   = blockIdx.y * 128;
  const int n0   = blockIdx.x * 128;

  const unsigned ldsA[2] = {(unsigned)(uintptr_t)&As[0][0][0],
                            (unsigned)(uintptr_t)&As[1][0][0]};
  const unsigned ldsB[2] = {(unsigned)(uintptr_t)&Bs[0][0][0],
                            (unsigned)(uintptr_t)&Bs[1][0][0]};

  const int nt = K >> 5;                // K / 32 tiles

  if (wid == 0) {                       // prologue: DMA tile 0 into buffer 0
    tdm_load_2d(A + (size_t)m0 * K, ldsA[0], (unsigned)K, (unsigned)M, 32, 128,
                (unsigned)K, 3, 3);
    tdm_load_2d(W + (size_t)n0 * K, ldsB[0], (unsigned)K, (unsigned)N, 32, 128,
                (unsigned)K, 3, 3);
  }

  f32x8 acc[2][4] = {};

  for (int t = 0; t < nt; ++t) {
    const int cur = t & 1;
    if (wid == 0) {
      if (t + 1 < nt) {                 // DMA next tile into the other buffer
        const int k1 = (t + 1) << 5;
        tdm_load_2d(A + (size_t)m0 * K + k1, ldsA[cur ^ 1], (unsigned)K,
                    (unsigned)M, 32, 128, (unsigned)K, 3, 3);
        tdm_load_2d(W + (size_t)n0 * K + k1, ldsB[cur ^ 1], (unsigned)K,
                    (unsigned)N, 32, 128, (unsigned)K, 3, 3);
        __builtin_amdgcn_s_wait_tensorcnt(2);   // oldest pair (tile t) done
      } else {
        __builtin_amdgcn_s_wait_tensorcnt(0);
      }
    }
    asm volatile("" ::: "memory");      // LDS written by TDM (address escaped)
    __syncthreads();                    // publish buffer `cur` to all waves

    bf16x16 af[2], bf[4];
#pragma unroll
    for (int mi = 0; mi < 2; ++mi)
      af[mi] = load_frag(&As[cur][wm * 32 + mi * 16][0], 40, lane);
#pragma unroll
    for (int ni = 0; ni < 4; ++ni)
      bf[ni] = load_frag(&Bs[cur][wn * 64 + ni * 16][0], 40, lane);
#pragma unroll
    for (int mi = 0; mi < 2; ++mi)
#pragma unroll
      for (int ni = 0; ni < 4; ++ni)
        acc[mi][ni] = WMMA_BF16(af[mi], bf[ni], acc[mi][ni]);

    __syncthreads();                    // all waves done before buffer reuse
  }

  // Epilogue. C layout: VGPR j -> row j (lanes 0-15) or row 8+j (lanes 16-31),
  // col = lane&15 within the 16-wide subtile.
  const int r0 = (lane >> 4) << 3;
  const int cl = lane & 15;
#pragma unroll
  for (int mi = 0; mi < 2; ++mi) {
#pragma unroll
    for (int ni = 0; ni < 4; ++ni) {
      const int col = n0 + wn * 64 + ni * 16 + cl;
      const float bv = HAS_BIAS ? bias[col] : 0.0f;
#pragma unroll
      for (int j = 0; j < 8; ++j) {
        const int row = m0 + wm * 32 + mi * 16 + r0 + j;
        float v = acc[mi][ni][j] + bv;
        if (GELU_ACT) v = 0.5f * v * (1.0f + erff(v * 0.70710678118f));
        if (RESIDUAL) v += res[(size_t)row * N + col];
        if (OUT_BF16) ((bf16_t*)outp)[(size_t)row * N + col] = (bf16_t)v;
        else          ((float*)outp)[(size_t)row * N + col]  = v;
      }
    }
  }
}

// ---------------------------------------------------------------------------
// Flash-attention: one workgroup per (b, h, 64-row query tile); 4 waves, each
// owning 16 query rows. Streaming softmax over 16 key tiles of 64.
// K tile staged by TDM (pad codes 4/3 -> 144B row stride); V tile transposed
// manually (TDM has no transpose); P transposed through per-wave LDS scratch.
// qkv: bf16 [B*N, 3D], Q at col h*64, K at 1024+h*64, V at 2048+h*64.
// ---------------------------------------------------------------------------
__global__ __launch_bounds__(128) void attention_kernel(const bf16_t* __restrict__ qkv,
                                                        bf16_t* __restrict__ atty) {
  __shared__ __align__(16) bf16_t Ks[64][72];       // key-major, hs contiguous
  __shared__ __align__(16) bf16_t VT[64][72];       // hs-major, key contiguous
  __shared__ __align__(16) bf16_t Ps[4][16][72];    // per-wave P scratch

  const int tid  = threadIdx.x;
  const int lane = tid & 31;
  const int wid  = tid >> 5;
  const int blk  = blockIdx.x;
  const int mt   = blk & 15;            // query tile
  const int bh   = blk >> 4;
  const int h    = bh & 15;
  const int b    = bh >> 4;

  const int    mrow0   = mt * 64 + wid * 16;
  const size_t rowbase = (size_t)b * NN;
  const unsigned ldsK  = (unsigned)(uintptr_t)&Ks[0][0];

  // Q fragments (16 rows x 64 hs) straight from global (row stride 3072 el).
  const bf16_t* qbase = qkv + (rowbase + mrow0) * 3072 + h * 64;
  const bf16x16 qf0 = load_frag(qbase,      3072, lane);
  const bf16x16 qf1 = load_frag(qbase + 32, 3072, lane);

  f32x8 o[4] = {};
  float mrow[8], lrow[8];
#pragma unroll
  for (int j = 0; j < 8; ++j) { mrow[j] = -1e30f; lrow[j] = 0.0f; }

  const int r0 = (lane >> 4) << 3;
  const int cl = lane & 15;

  for (int kt = 0; kt < 16; ++kt) {
    __syncthreads();                    // prior tile fully consumed
    // K tile 64x64 via TDM: tensor row stride 3072, LDS rows padded to 144B.
    if (wid == 0)
      tdm_load_2d(qkv + (rowbase + kt * 64) * 3072 + 1024 + h * 64, ldsK,
                  3072u, (unsigned)MTOT, 64, 64, 3072u, 4, 3);
    // V staged transposed so the P*V B-operand has keys contiguous per lane.
#pragma unroll
    for (int i = 0; i < 32; ++i) {
      int idx = tid + i * 128;          // 0..4095
      int key = idx >> 6;
      int hs  = idx & 63;
      VT[hs][key] = qkv[(rowbase + kt * 64 + key) * 3072 + 2048 + h * 64 + hs];
    }
    if (wid == 0) __builtin_amdgcn_s_wait_tensorcnt(0);
    asm volatile("" ::: "memory");
    __syncthreads();

    // S = Q K^T (16 x 64), K-dim 64 => two chained WMMAs per 16x16 subtile.
    f32x8 s[4];
#pragma unroll
    for (int ni = 0; ni < 4; ++ni) {
      f32x8 z = {};
      bf16x16 kb0 = load_frag(&Ks[ni * 16][0],  72, lane);
      bf16x16 kb1 = load_frag(&Ks[ni * 16][32], 72, lane);
      z = WMMA_BF16(qf0, kb0, z);
      z = WMMA_BF16(qf1, kb1, z);
      s[ni] = z;
    }

    // Online softmax in the C layout; row reductions over 16-lane halves.
#pragma unroll
    for (int j = 0; j < 8; ++j) {
      float mx = -1e30f;
#pragma unroll
      for (int ni = 0; ni < 4; ++ni) {
        float v = s[ni][j] * 0.125f;    // HS^-0.5
        s[ni][j] = v;
        mx = fmaxf(mx, v);
      }
#pragma unroll
      for (int off = 1; off < 16; off <<= 1) mx = fmaxf(mx, __shfl_xor(mx, off, 32));
      const float mnew  = fmaxf(mrow[j], mx);
      const float alpha = __expf(mrow[j] - mnew);
      mrow[j] = mnew;
      float rs = 0.0f;
#pragma unroll
      for (int ni = 0; ni < 4; ++ni) {
        float p = __expf(s[ni][j] - mnew);
        s[ni][j] = p;
        rs += p;
      }
#pragma unroll
      for (int off = 1; off < 16; off <<= 1) rs += __shfl_xor(rs, off, 32);
      lrow[j] = lrow[j] * alpha + rs;
#pragma unroll
      for (int ni = 0; ni < 4; ++ni) o[ni][j] *= alpha;
    }

    // C-layout -> A-layout for P via per-wave LDS scratch (DS ops in-order).
#pragma unroll
    for (int ni = 0; ni < 4; ++ni)
#pragma unroll
      for (int j = 0; j < 8; ++j)
        Ps[wid][r0 + j][ni * 16 + cl] = (bf16_t)s[ni][j];

    bf16x16 pa0 = load_frag(&Ps[wid][0][0],  72, lane);
    bf16x16 pa1 = load_frag(&Ps[wid][0][32], 72, lane);

    // O += P * V  (B operand rows = VT rows: hs-major, keys contiguous).
#pragma unroll
    for (int ni = 0; ni < 4; ++ni) {
      bf16x16 vb0 = load_frag(&VT[ni * 16][0],  72, lane);
      bf16x16 vb1 = load_frag(&VT[ni * 16][32], 72, lane);
      o[ni] = WMMA_BF16(pa0, vb0, o[ni]);
      o[ni] = WMMA_BF16(pa1, vb1, o[ni]);
    }
  }

  // Normalize and emit bf16 activations for the projection GEMM.
#pragma unroll
  for (int ni = 0; ni < 4; ++ni) {
#pragma unroll
    for (int j = 0; j < 8; ++j) {
      const float inv = 1.0f / lrow[j];
      atty[(rowbase + mrow0 + r0 + j) * DD + h * 64 + ni * 16 + cl] =
          (bf16_t)(o[ni][j] * inv);
    }
  }
}

// ---------------------------------------------------------------------------
// Host orchestration (all on `stream`, graph-capture safe).
// ---------------------------------------------------------------------------
static inline size_t align256(size_t x) { return (x + 255) & ~(size_t)255; }

extern "C" void kernel_launch(void* const* d_in, const int* in_sizes, int n_in,
                              void* d_out, int out_size, void* d_ws, size_t ws_size,
                              hipStream_t stream) {
  const float* x      = (const float*)d_in[0];
  const float* qkv_w  = (const float*)d_in[1];
  const float* proj_w = (const float*)d_in[2];
  const float* proj_b = (const float*)d_in[3];
  const float* ln1_w  = (const float*)d_in[4];
  const float* ln1_b  = (const float*)d_in[5];
  const float* fc1_w  = (const float*)d_in[6];
  const float* fc1_b  = (const float*)d_in[7];
  const float* fc2_w  = (const float*)d_in[8];
  const float* fc2_b  = (const float*)d_in[9];
  const float* ln2_w  = (const float*)d_in[10];
  const float* ln2_b  = (const float*)d_in[11];
  float* out = (float*)d_out;

  // Workspace carve-up (bf16 weights, bf16 activations, fp32 residual x1).
  char* ws = (char*)d_ws;
  size_t off = 0;
  bf16_t* w_qkv  = (bf16_t*)(ws + off); off = align256(off + (size_t)3 * DD * DD * 2);
  bf16_t* w_proj = (bf16_t*)(ws + off); off = align256(off + (size_t)DD * DD * 2);
  bf16_t* w_fc1  = (bf16_t*)(ws + off); off = align256(off + (size_t)HIDD * DD * 2);
  bf16_t* w_fc2  = (bf16_t*)(ws + off); off = align256(off + (size_t)DD * HIDD * 2);
  bf16_t* ln_act = (bf16_t*)(ws + off); off = align256(off + (size_t)MTOT * DD * 2);
  bf16_t* qkvact = (bf16_t*)(ws + off); off = align256(off + (size_t)MTOT * 3 * DD * 2);
  bf16_t* atty   = (bf16_t*)(ws + off); off = align256(off + (size_t)MTOT * DD * 2);
  float*  x1     = (float*)(ws + off);  off = align256(off + (size_t)MTOT * DD * 4);
  bf16_t* hbuf   = (bf16_t*)(ws + off); off = align256(off + (size_t)MTOT * HIDD * 2);
  (void)ws_size; (void)n_in; (void)in_sizes; (void)out_size;

  // 1) Weights -> bf16.
  cvt_f32_to_bf16<<<(3 * DD * DD + 255) / 256, 256, 0, stream>>>(qkv_w, w_qkv, 3 * DD * DD);
  cvt_f32_to_bf16<<<(DD * DD + 255) / 256, 256, 0, stream>>>(proj_w, w_proj, DD * DD);
  cvt_f32_to_bf16<<<(HIDD * DD + 255) / 256, 256, 0, stream>>>(fc1_w, w_fc1, HIDD * DD);
  cvt_f32_to_bf16<<<(DD * HIDD + 255) / 256, 256, 0, stream>>>(fc2_w, w_fc2, DD * HIDD);

  // 2) ln1(x) -> bf16.
  layernorm_to_bf16<<<MTOT, 256, 0, stream>>>(x, ln1_w, ln1_b, ln_act);

  // 3) qkv = ln1 @ qkv_w^T   [4096 x 3072], bf16 out.
  gemm_bf16_kernel<false, false, false, true>
      <<<dim3(3 * DD / 128, MTOT / 128), 256, 0, stream>>>(
          ln_act, w_qkv, nullptr, nullptr, qkvact, MTOT, 3 * DD, DD);

  // 4) attention -> atty (bf16 [4096 x 1024]).
  attention_kernel<<<BB * NHH * (NN / 64), 128, 0, stream>>>(qkvact, atty);

  // 5) x1 = x + atty @ proj_w^T + proj_b   (fp32).
  gemm_bf16_kernel<true, false, true, false>
      <<<dim3(DD / 128, MTOT / 128), 256, 0, stream>>>(
          atty, w_proj, proj_b, x, x1, MTOT, DD, DD);

  // 6) ln2(x1) -> bf16 (reuses ln_act; prior consumer finished).
  layernorm_to_bf16<<<MTOT, 256, 0, stream>>>(x1, ln2_w, ln2_b, ln_act);

  // 7) h = gelu(ln2 @ fc1_w^T + fc1_b)  (bf16 [4096 x 4096]).
  gemm_bf16_kernel<true, true, false, true>
      <<<dim3(HIDD / 128, MTOT / 128), 256, 0, stream>>>(
          ln_act, w_fc1, fc1_b, nullptr, hbuf, MTOT, HIDD, DD);

  // 8) out = x1 + h @ fc2_w^T + fc2_b  (fp32).
  gemm_bf16_kernel<true, false, true, false>
      <<<dim3(DD / 128, MTOT / 128), 256, 0, stream>>>(
          hbuf, w_fc2, fc2_b, x1, out, MTOT, DD, HIDD);
}
